// AdaptConv2d_34479997452483
// MI455X (gfx1250) — matrix operand order repaired and checked by reading, not compiled, not run on hardware
//
#include <hip/hip_runtime.h>
#include <hip/hip_bf16.h>
#include <math.h>

// ---------------------------------------------------------------------------
// Types for CDNA5 WMMA (wave32, 16x16x32 bf16 -> f32)
// ---------------------------------------------------------------------------
typedef __bf16 bf16_t;
typedef __attribute__((ext_vector_type(2)))  __bf16 v2bf;
typedef __attribute__((ext_vector_type(16))) __bf16 v16bf;
typedef __attribute__((ext_vector_type(8)))  float  v8f;
typedef __attribute__((ext_vector_type(4)))  unsigned int v4u;

__device__ inline float sigm(float x) { return 1.0f / (1.0f + expf(-x)); }

// Problem constants
#define CC   256
#define IHW  56
#define HWSZ 3136   // 56*56
#define WELEM (9 * 8 * 256 * 32)  // packed weight elements per conv
#define NKS  72                   // k-steps: 9 taps * 8 ci-blocks

// ---------------------------------------------------------------------------
// Weight pre-pack: fp32 OIHW [co][ci][3][3] -> bf16 [ks][co][p], where ks =
// tap*8 + ci_block and p = k with bits 3<->4 swapped (involution). Each
// lane's A-matrix fragment (K chunks {0-7,16-23} / {8-15,24-31}, ISA 7.12.2)
// becomes a contiguous 32B run -> bare ds_load_b128 pairs, no VALU repacking,
// and the per-k-step global base is linear (+8192 elements).
// ---------------------------------------------------------------------------
__global__ void pack_weights(const float* __restrict__ w,
                             bf16_t* __restrict__ wt) {
  const int idx = blockIdx.x * 256 + threadIdx.x;  // < WELEM
  const int p   = idx & 31;
  const int co  = (idx >> 5) & 255;
  const int cib = (idx >> 13) & 7;
  const int r   = idx >> 16;
  const int kk  = (p & 7) | ((p & 8) << 1) | ((p & 16) >> 1);  // swap bits 3,4
  const int ci  = cib * 32 + kk;
  wt[idx] = (bf16_t)w[(co * CC + ci) * 9 + r];
}

// ---------------------------------------------------------------------------
// Implicit-GEMM 3x3 conv with bf16 WMMA, fp32 accumulate.
//   GEMM: D[co][m] = sum_k W[co][k] * P[k][m],  k = tap*256 + ci
//   A (SRC0) = weights 16x32 (pre-packed bf16), B (SRC1) = patches 32x16
// Block: 256 threads (8 waves). Tile: 128 co x 128 positions, K-step 32.
// Each wave owns 32 co x 64 pos -> 2x4 accumulators -> 8 WMMA per k-step.
// Ping-pong LDS staging: one __syncthreads per k-step; tile ks+1 streams in
// (global loads + ds stores) while WMMAs consume tile ks.
// MODE 0: main conv, fused mask/skip blend into NCHW output.
// MODE 1: gate conv, fused bias + ReLU + spatial-mean (atomic) into g[n][co].
// ---------------------------------------------------------------------------
template <int STRIDE, int PAD, int OW, int OHW, int MODE>
__global__ __launch_bounds__(256) void conv_wmma(
    const float* __restrict__ x, const bf16_t* __restrict__ wt,
    const float* __restrict__ bias, const float* __restrict__ mask,
    const float* __restrict__ skip, float* __restrict__ outp, int Mpos) {
  constexpr int BM = 128;   // co tile
  constexpr int BN = 128;   // position tile
  constexpr int LDP = 40;   // padded LDS row (bf16) = 80B; 16B-aligned chunks

  __shared__ bf16_t Ws[2][BM][LDP];   // weights [co][p] (fragment order)
  __shared__ bf16_t Xs[2][BN][LDP];   // patches [m][p]  (fragment order, k bits 3<->4 swapped)

  const int t    = threadIdx.x;
  const int lane = t & 31;
  const int wv   = t >> 5;
  const int wr   = wv & 3;    // co subtile  (4 x 32)
  const int wc   = wv >> 2;   // pos subtile (2 x 64)

  const int m0  = blockIdx.x * BN;
  const int co0 = blockIdx.y * BM;

  // --- patch loader role: thread owns row m0+(t&127), ci-pairs 4i+2h ---
  const int mrow = t & 127;
  const int h    = t >> 7;
  const int m    = m0 + mrow;
  const bool mvalid = (m < Mpos);
  int n = 0, oh = 0, ow = 0;
  if (mvalid) {
    n = m / OHW;
    int hw = m - n * OHW;
    oh = hw / OW;
    ow = hw - oh * OW;
  }
  // per-tap source offsets + validity bitmask (hoisted out of the hot loop)
  int xoffs[9];
  unsigned pmask = 0;
#pragma unroll
  for (int r = 0; r < 9; ++r) {
    const int dh = r / 3, dw = r - dh * 3;
    const int ih = oh * STRIDE + dh - PAD;
    const int iw = ow * STRIDE + dw - PAD;
    const bool pv =
        mvalid && ((unsigned)ih < (unsigned)IHW) && ((unsigned)iw < (unsigned)IHW);
    pmask |= (pv ? 1u : 0u) << r;
    xoffs[r] = n * (CC * HWSZ) + ih * IHW + iw;
  }

  // --- weight loader role: thread copies 32B: co row t>>1, half t&1 ---
  const int wnl = t >> 1;
  const int whh = t & 1;
  const bf16_t* wtt = wt + (co0 + wnl) * 32 + whh * 16;  // + ks*8192

  // stage tile `s` into buffer s&1
  auto stage = [&](int s) {
    const int buf = s & 1;
    {
      const v4u* src = (const v4u*)(wtt + s * 8192);
      v4u w0 = src[0];
      v4u w1 = src[1];
      *(v4u*)&Ws[buf][wnl][whh * 16]     = w0;
      *(v4u*)&Ws[buf][wnl][whh * 16 + 8] = w1;
    }
    const int r   = s >> 3;
    const int ci0 = (s & 7) * 32;
    const bool pv = (pmask >> r) & 1u;
    const int xo  = xoffs[r];
#pragma unroll
    for (int i = 0; i < 8; ++i) {
      const int cl = 4 * i + 2 * h;
      float xa = 0.0f, xb = 0.0f;
      if (pv) {
        const float* px = x + xo + (ci0 + cl) * HWSZ;
        xa = px[0];
        xb = px[HWSZ];
      }
      v2bf st;
      st[0] = (bf16_t)xa;
      st[1] = (bf16_t)xb;
      *(v2bf*)&Xs[buf][mrow][(cl & 7) | ((cl & 8) << 1) | ((cl & 16) >> 1)] = st;
    }
  };

  v8f acc[2][4];
#pragma unroll
  for (int a = 0; a < 2; ++a)
#pragma unroll
    for (int b = 0; b < 4; ++b)
#pragma unroll
      for (int e = 0; e < 8; ++e) acc[a][b][e] = 0.0f;

  const int l16 = lane & 15;
  const int p0  = lane & 16;

  stage(0);  // prologue
  for (int ks = 0; ks < NKS; ++ks) {
    const int cur = ks & 1;
    __syncthreads();          // staging of tile ks complete (all waves)
    if (ks + 1 < NKS) stage(ks + 1);  // overlap: stream tile ks+1
    if (ks + 2 < NKS) {               // prefetch patch slice 2 steps ahead
      const int r2 = (ks + 2) >> 3;
      if ((pmask >> r2) & 1u)
        __builtin_prefetch(x + xoffs[r2] + (((ks + 2) & 7) * 32 + 2 * h) * HWSZ,
                           0, 3);
    }

    // ---- fragments: contiguous 32B per lane (2x ds_load_b128 each) ----
    v16bf afrag[2], bfrag[4];
#pragma unroll
    for (int rt = 0; rt < 2; ++rt)
      __builtin_memcpy(&afrag[rt], &Ws[cur][wr * 32 + rt * 16 + l16][p0], 32);
#pragma unroll
    for (int ct = 0; ct < 4; ++ct)
      __builtin_memcpy(&bfrag[ct], &Xs[cur][wc * 64 + ct * 16 + l16][p0], 32);

#pragma unroll
    for (int rt = 0; rt < 2; ++rt)
#pragma unroll
      for (int ct = 0; ct < 4; ++ct)
        acc[rt][ct] = __builtin_amdgcn_wmma_f32_16x16x32_bf16(
            false, afrag[rt], false, bfrag[ct], (short)0, acc[rt][ct],
            false, false);
  }

  // ---- epilogue: D rows = co (VGPR idx), cols = positions (lanes) ----
#pragma unroll
  for (int ct = 0; ct < 4; ++ct) {
    const int mm = m0 + wc * 64 + ct * 16 + l16;
    if (mm >= Mpos) continue;
    const int n2 = mm / OHW;
    if constexpr (MODE == 0) {
      const int hw = mm - n2 * OHW;
      const float sk = skip[n2];
#pragma unroll
      for (int rt = 0; rt < 2; ++rt) {
        const int cob = co0 + wr * 32 + rt * 16 + ((lane < 16) ? 0 : 8);
        int addr = (n2 * CC + cob) * OHW + hw;
#pragma unroll
        for (int g = 0; g < 8; ++g) {
          const float xv = x[addr];
          const float mk = mask[n2 * CC + cob + g];
          outp[addr] = (sk != 0.0f || mk == 0.0f) ? xv : acc[rt][ct][g];
          addr += OHW;
        }
      }
    } else {
#pragma unroll
      for (int rt = 0; rt < 2; ++rt) {
        const int cob = co0 + wr * 32 + rt * 16 + ((lane < 16) ? 0 : 8);
#pragma unroll
        for (int g = 0; g < 8; ++g) {
          float v = acc[rt][ct][g] + bias[cob + g];
          v = fmaxf(v, 0.0f);
          atomicAdd(&outp[n2 * CC + cob + g], v * (1.0f / (float)OHW));
        }
      }
    }
  }
}

// ---------------------------------------------------------------------------
// Small helper kernels (gate math: negligible FLOPs, plain VALU)
// ---------------------------------------------------------------------------
__global__ void zero_kernel(float* __restrict__ p, int nelem) {
  int i = blockIdx.x * blockDim.x + threadIdx.x;
  if (i < nelem) p[i] = 0.0f;
}

__global__ void avgpool_kernel(const float* __restrict__ x,
                               float* __restrict__ g2) {
  const int nc = blockIdx.x;  // n*256 + c
  const float* p = x + (size_t)nc * HWSZ;
  float s = 0.0f;
  for (int i = threadIdx.x; i < HWSZ; i += 256) s += p[i];
  __shared__ float red[256];
  red[threadIdx.x] = s;
  __syncthreads();
  for (int off = 128; off > 0; off >>= 1) {
    if (threadIdx.x < off) red[threadIdx.x] += red[threadIdx.x + off];
    __syncthreads();
  }
  if (threadIdx.x == 0) g2[nc] = red[0] * (1.0f / (float)HWSZ);
}

__global__ void layergate_kernel(const float* __restrict__ g2,
                                 const float* __restrict__ lgw,
                                 const float* __restrict__ lgb,
                                 const float* __restrict__ wih,
                                 const float* __restrict__ bih,
                                 const float* __restrict__ bhh,
                                 const float* __restrict__ fw,
                                 const float* __restrict__ fb,
                                 float* __restrict__ layerbit) {
  const int n = threadIdx.x;
  if (n >= 32) return;
  float hbuf[10];
#pragma unroll
  for (int j = 0; j < 10; ++j) {
    float s = lgb[j];
    for (int c = 0; c < CC; ++c) s += lgw[j * CC + c] * g2[n * CC + c];
    hbuf[j] = fmaxf(s, 0.0f);
  }
  float o = fb[0];
#pragma unroll
  for (int j = 0; j < 10; ++j) {
    float gi = bih[j] + bhh[j];
    float gg = bih[20 + j] + bhh[20 + j];
    float go = bih[30 + j] + bhh[30 + j];
#pragma unroll
    for (int k = 0; k < 10; ++k) {
      gi += wih[j * 10 + k] * hbuf[k];
      gg += wih[(20 + j) * 10 + k] * hbuf[k];
      go += wih[(30 + j) * 10 + k] * hbuf[k];
    }
    const float cc = sigm(gi) * tanhf(gg);
    const float hs = sigm(go) * tanhf(cc);
    o += fw[j] * hs;
  }
  o = fmaxf(o, 0.0f);
  layerbit[n] = rintf(sigm(o));  // round-half-even, matches jnp.round
}

__global__ void channelfc_kernel(const float* __restrict__ gbuf,
                                 const float* __restrict__ fcw,
                                 const float* __restrict__ fcb,
                                 const float* __restrict__ layerbit,
                                 float* __restrict__ mask,
                                 float* __restrict__ skip) {
  const int n = blockIdx.x, c = threadIdx.x;
  __shared__ float gs[256];
  gs[c] = gbuf[n * CC + c];
  __syncthreads();
  float s = fcb[c];
  for (int k = 0; k < CC; ++k) s += fcw[c * CC + k] * gs[k];
  s = fmaxf(s, 0.0f);
  const float mk = rintf(sigm(s));
  mask[n * CC + c] = mk;
  __shared__ float red[256];
  red[c] = mk;
  __syncthreads();
  for (int off = 128; off > 0; off >>= 1) {
    if (c < off) red[c] += red[c + off];
    __syncthreads();
  }
  if (c == 0)
    skip[n] = (layerbit[n] == 0.0f || red[0] == 0.0f) ? 1.0f : 0.0f;
}

// ---------------------------------------------------------------------------
// Launch
// ---------------------------------------------------------------------------
extern "C" void kernel_launch(void* const* d_in, const int* in_sizes, int n_in,
                              void* d_out, int out_size, void* d_ws,
                              size_t ws_size, hipStream_t stream) {
  (void)in_sizes; (void)n_in; (void)out_size; (void)ws_size;
  const float* x      = (const float*)d_in[0];
  const float* conv_w = (const float*)d_in[1];
  const float* cgw    = (const float*)d_in[2];
  const float* cgb    = (const float*)d_in[3];
  const float* fcw    = (const float*)d_in[4];
  const float* fcb    = (const float*)d_in[5];
  const float* lgw    = (const float*)d_in[6];
  const float* lgb    = (const float*)d_in[7];
  const float* wih    = (const float*)d_in[8];
  /* d_in[9] = lstm_w_hh: unused by reference math (only its bias enters) */
  const float* bih    = (const float*)d_in[10];
  const float* bhh    = (const float*)d_in[11];
  const float* lfw    = (const float*)d_in[12];
  const float* lfb    = (const float*)d_in[13];

  float* out = (float*)d_out;
  float* ws  = (float*)d_ws;
  float* gbuf     = ws;                       // 8192: mean(relu(gate conv))
  float* g2       = ws + 8192;                // 8192: global avg pool
  float* layerbit = ws + 16384;               // 32 (padded to 64)
  float* mask     = ws + 16384 + 64;          // 8192
  float* skip     = ws + 16384 + 64 + 8192;   // 32 (padded to 64)
  bf16_t* wt_main = (bf16_t*)(ws + 16384 + 64 + 8192 + 64);
  bf16_t* wt_gate = wt_main + WELEM;

  // one-time weight packing (bf16, fragment-ordered, coalesced)
  pack_weights<<<WELEM / 256, 256, 0, stream>>>(conv_w, wt_main);
  pack_weights<<<WELEM / 256, 256, 0, stream>>>(cgw, wt_gate);

  zero_kernel<<<32, 256, 0, stream>>>(gbuf, 32 * CC);
  avgpool_kernel<<<32 * CC, 256, 0, stream>>>(x, g2);
  layergate_kernel<<<1, 32, 0, stream>>>(g2, lgw, lgb, wih, bih, bhh, lfw, lfb,
                                         layerbit);
  // channel-gate conv: stride 2, pad 0, 27x27 output, fused ReLU+mean
  conv_wmma<2, 0, 27, 729, 1><<<dim3(183, 2), 256, 0, stream>>>(
      x, wt_gate, cgb, nullptr, nullptr, gbuf, 32 * 729);
  channelfc_kernel<<<32, 256, 0, stream>>>(gbuf, fcw, fcb, layerbit, mask,
                                           skip);
  // main conv: stride 1, pad 1, fused mask/skip blend
  conv_wmma<1, 1, 56, 3136, 0><<<dim3(784, 2), 256, 0, stream>>>(
      x, wt_main, nullptr, mask, skip, out, 32 * 3136);
}
